// SocialLSTM_17033840296106
// MI455X (gfx1250) — matrix operand order repaired and checked by reading, not compiled, not run on hardware
//
#include <hip/hip_runtime.h>

// ---------------------------------------------------------------------------
// Social-LSTM fused forward for MI455X (gfx1250, wave32, WMMA).
//   * 1 wave == 16 agents == one WMMA M-tile; h,c held in C-layout v8f regs.
//   * gates GEMM  : [16,64]x[64,256] -> 16 ntiles x 2 kchunks WMMA f16.
//   * social GEMM : per-row W1-block selection done by per-lane A-masking
//                   (lane L holds only row L&15 of the A fragment).
//   * weights staged once per block into f16 B-fragment layout in LDS.
//   * all sigmoid/tanh via v_exp_f32 + v_rcp_f32 (no IEEE div sequences).
// ---------------------------------------------------------------------------

typedef __attribute__((ext_vector_type(16))) _Float16 v16h;
typedef __attribute__((ext_vector_type(8)))  float    v8f;

#define HID    64
#define GXC    4
#define GYC    4
#define GCELLS 16
#define TSTEPS 14
#define BATCH  65536
#define WAVES  4
#define BLOCK_THREADS (WAVES * 32)
#define AGENTS_PER_BLOCK (WAVES * 16)

// ---- LDS layout (byte offsets; all 16B aligned) ----
#define OFF_W1F    0                         // 128 frags * 512 halfs = 131072 B
#define OFF_WHHF   (OFF_W1F   + 131072)      //  32 frags             =  32768 B
#define OFF_W2F    (OFF_WHHF  + 32768)       //   8 frags             =   8192 B
#define OFF_WOUTF  (OFF_W2F   + 8192)        //   2 frags             =   2048 B
#define OFF_BSUM   (OFF_WOUTF + 2048)        // 256 f32
#define OFF_WC0    (OFF_BSUM  + 1024)        // 256 f32
#define OFF_WC1    (OFF_WC0   + 1024)        // 256 f32
#define OFF_B1     (OFF_WC1   + 1024)        //  64 f32
#define OFF_B2     (OFF_B1    + 256)         //  64 f32
#define OFF_BOUT   (OFF_B2    + 256)         //   4 f32 (2 used)
#define OFF_HTILE  (OFF_BOUT  + 16)          // WAVES * 16*64 halfs   =   8192 B
#define OFF_POST   (OFF_HTILE + 8192)        // WAVES * 32 f32        =    512 B
#define LDS_BYTES  (OFF_POST  + 512)

union V16 { v16h v; unsigned u[8]; uint4 q[2]; };

__device__ __forceinline__ float fastrcp(float x) {
  return __builtin_amdgcn_rcpf(x);           // v_rcp_f32
}
__device__ __forceinline__ float sigm(float x) {
  return fastrcp(1.0f + __expf(-x));         // saturates correctly at +-inf
}
__device__ __forceinline__ float tanhfast(float x) {
  // tanh(x) = 1 - 2/(1+e^(2x)); e->inf => 1, e->0 => -1 (self-saturating)
  float e = __expf(2.0f * x);
  return __builtin_fmaf(-2.0f, fastrcp(1.0f + e), 1.0f);
}

// monotone float<->uint key for atomic min/max
__device__ __forceinline__ unsigned fkey(float f) {
  unsigned u = __float_as_uint(f);
  return (u & 0x80000000u) ? ~u : (u | 0x80000000u);
}
__device__ __forceinline__ float ikey(unsigned k) {
  unsigned u = (k & 0x80000000u) ? (k ^ 0x80000000u) : ~k;
  return __uint_as_float(u);
}

// B-fragment load: 16 contiguous halfs per lane (2x ds_load_b128)
__device__ __forceinline__ v16h ldsFrag(const _Float16* base, int frag, int lane) {
  V16 r;
  const uint4* p = (const uint4*)(base + frag * 512 + lane * 16);
  r.q[0] = p[0];
  r.q[1] = p[1];
  return r.v;
}

// A-layout gather from a 16x64 f16 row-major tile (pairs are contiguous)
__device__ __forceinline__ void loadA(const _Float16* ht, int row, int sel,
                                      V16& lo, V16& hi) {
#pragma unroll
  for (int v = 0; v < 8; ++v) {
    int base = v * 2 + sel * 16;
    lo.u[v] = *(const unsigned*)(ht + row * 64 + base);
    hi.u[v] = *(const unsigned*)(ht + row * 64 + 32 + base);
  }
}

// ---------------------------------------------------------------------------
__global__ void minmax_init(unsigned* ws) {
  int i = threadIdx.x;
  if (i < TSTEPS * 4) ws[i] = (i & 1) ? 0u : 0xFFFFFFFFu;
}

__global__ void __launch_bounds__(256)
minmax_reduce(const float* __restrict__ traj, unsigned* __restrict__ ws) {
  int t = blockIdx.y;
  __shared__ unsigned s[4];
  if (threadIdx.x < 4) s[threadIdx.x] = (threadIdx.x & 1) ? 0u : 0xFFFFFFFFu;
  __syncthreads();
  unsigned mnx = 0xFFFFFFFFu, mxx = 0u, mny = 0xFFFFFFFFu, mxy = 0u;
  for (int b = blockIdx.x * 256 + threadIdx.x; b < BATCH; b += gridDim.x * 256) {
    float x = traj[((size_t)t * BATCH + b) * 2 + 0];
    float y = traj[((size_t)t * BATCH + b) * 2 + 1];
    unsigned kx = fkey(x), ky = fkey(y);
    mnx = min(mnx, kx); mxx = max(mxx, kx);
    mny = min(mny, ky); mxy = max(mxy, ky);
  }
  atomicMin(&s[0], mnx); atomicMax(&s[1], mxx);
  atomicMin(&s[2], mny); atomicMax(&s[3], mxy);
  __syncthreads();
  if (threadIdx.x < 4) {
    if (threadIdx.x & 1) atomicMax(&ws[t * 4 + threadIdx.x], s[threadIdx.x]);
    else                 atomicMin(&ws[t * 4 + threadIdx.x], s[threadIdx.x]);
  }
}

// ---------------------------------------------------------------------------
__global__ void __launch_bounds__(BLOCK_THREADS)
social_lstm_main(const float* __restrict__ traj,
                 const float* __restrict__ Wih, const float* __restrict__ Whh,
                 const float* __restrict__ bih, const float* __restrict__ bhh,
                 const float* __restrict__ W1g, const float* __restrict__ b1g,
                 const float* __restrict__ W2g, const float* __restrict__ b2g,
                 const float* __restrict__ Woutg, const float* __restrict__ boutg,
                 const unsigned* __restrict__ mmkeys,
                 float* __restrict__ out) {
  extern __shared__ char smem[];
  _Float16* w1f   = (_Float16*)(smem + OFF_W1F);
  _Float16* whhf  = (_Float16*)(smem + OFF_WHHF);
  _Float16* w2f   = (_Float16*)(smem + OFF_W2F);
  _Float16* woutf = (_Float16*)(smem + OFF_WOUTF);
  float* bsum  = (float*)(smem + OFF_BSUM);
  float* wc0   = (float*)(smem + OFF_WC0);
  float* wc1   = (float*)(smem + OFF_WC1);
  float* b1s   = (float*)(smem + OFF_B1);
  float* b2s   = (float*)(smem + OFF_B2);
  float* bouts = (float*)(smem + OFF_BOUT);

  const int tid = threadIdx.x;

  // ---- one-time weight staging into f16 B-fragment layout ----
  // B order used here, same k order as loadA (16x16x32 f16): lane holds column n=lane&15, K = (lane>>4)*16 + e
  for (int i = tid; i < 32 * 512; i += BLOCK_THREADS) {       // W_hh^T tiles
    int frag = i >> 9, elem = i & 511;
    int ls = elem >> 4, e = elem & 15;
    int nl = ls & 15, kl = (ls >> 4) * 16 + e;
    int nt = frag >> 1, kc = frag & 1;
    whhf[i] = (_Float16)Whh[(nt * 16 + nl) * HID + kc * 32 + kl];
  }
  for (int i = tid; i < 128 * 512; i += BLOCK_THREADS) {      // W1 per-cell blocks
    int frag = i >> 9, elem = i & 511;
    int ls = elem >> 4, e = elem & 15;
    int nl = ls & 15, kl = (ls >> 4) * 16 + e;
    int cell = frag >> 3, jt = (frag >> 1) & 3, kc = frag & 1;
    w1f[i] = (_Float16)W1g[(jt * 16 + nl) * (GCELLS * HID) + cell * HID + kc * 32 + kl];
  }
  for (int i = tid; i < 8 * 512; i += BLOCK_THREADS) {        // W2^T tiles
    int frag = i >> 9, elem = i & 511;
    int ls = elem >> 4, e = elem & 15;
    int nl = ls & 15, kl = (ls >> 4) * 16 + e;
    int jt = frag >> 1, kc = frag & 1;
    w2f[i] = (_Float16)W2g[(jt * 16 + nl) * HID + kc * 32 + kl];
  }
  for (int i = tid; i < 2 * 512; i += BLOCK_THREADS) {        // Wout^T (padded N)
    int frag = i >> 9, elem = i & 511;
    int ls = elem >> 4, e = elem & 15;
    int nl = ls & 15, kl = (ls >> 4) * 16 + e;
    woutf[i] = (nl < 2) ? (_Float16)Woutg[nl * HID + frag * 32 + kl] : (_Float16)0.0f;
  }
  for (int i = tid; i < 256; i += BLOCK_THREADS) {
    bsum[i] = bih[i] + bhh[i];
    wc0[i] = Wih[i * 2 + 0];
    wc1[i] = Wih[i * 2 + 1];
  }
  for (int i = tid; i < 64; i += BLOCK_THREADS) { b1s[i] = b1g[i]; b2s[i] = b2g[i]; }
  if (tid < 2) bouts[tid] = boutg[tid];
  __syncthreads();

  // ---- per-wave tile setup ----
  const int wv = tid >> 5, lane = tid & 31;
  const int row = lane & 15, sel = lane >> 4, n15 = lane & 15;
  const int agent0 = blockIdx.x * AGENTS_PER_BLOCK + wv * 16;
  _Float16* ht = (_Float16*)(smem + OFF_HTILE) + wv * 1024;
  float* post = (float*)(smem + OFF_POST) + wv * 32;

  v8f cst[4], hC[4];
  V16 aLo, aHi;
#pragma unroll
  for (int jt = 0; jt < 4; ++jt)
#pragma unroll
    for (int r = 0; r < 8; ++r) { cst[jt][r] = 0.0f; hC[jt][r] = 0.0f; }
#pragma unroll
  for (int i = 0; i < 8; ++i) { aLo.u[i] = 0u; aHi.u[i] = 0u; }

#pragma unroll 1
  for (int t = 0; t < TSTEPS; ++t) {
    // stage this wave's 16 positions (coalesced 128B load per wave)
    {
      int m = lane >> 1, c = lane & 1;
      post[m * 2 + c] = traj[((size_t)t * BATCH + agent0 + m) * 2 + c];
    }
    asm volatile("s_wait_dscnt 0" ::: "memory");

    float px[8], py[8];
#pragma unroll
    for (int r = 0; r < 8; ++r) {
      int m = r + sel * 8;
      px[r] = post[m * 2 + 0];
      py[r] = post[m * 2 + 1];
    }
    float rx = post[row * 2 + 0], ry = post[row * 2 + 1];
    float mnx = ikey(mmkeys[t * 4 + 0]), mxx = ikey(mmkeys[t * 4 + 1]);
    float mny = ikey(mmkeys[t * 4 + 2]), mxy = ikey(mmkeys[t * 4 + 3]);
    int cx = (int)((rx - mnx) * ((float)GXC * fastrcp(mxx - mnx)));
    int cy = (int)((ry - mny) * ((float)GYC * fastrcp(mxy - mny)));
    cx = min(max(cx, 0), GXC - 1);
    cy = min(max(cy, 0), GYC - 1);
    int myidx = cy * GXC + cx;

    // ---- gates GEMM (h_{t-1} in A frags) + LSTM update ----
#pragma unroll
    for (int jt = 0; jt < 4; ++jt) {
      v8f ga[4];
#pragma unroll
      for (int g = 0; g < 4; ++g) {
        int nt = g * 4 + jt;
        int n = nt * 16 + n15;
        float bs = bsum[n], w0 = wc0[n], w1v = wc1[n];
        v8f a;
#pragma unroll
        for (int r = 0; r < 8; ++r) a[r] = bs + px[r] * w0 + py[r] * w1v;
        a = __builtin_amdgcn_wmma_f32_16x16x32_f16(
            false, aLo.v, false, ldsFrag(whhf, nt * 2 + 0, lane), (short)0, a, false, false);
        a = __builtin_amdgcn_wmma_f32_16x16x32_f16(
            false, aHi.v, false, ldsFrag(whhf, nt * 2 + 1, lane), (short)0, a, false, false);
        ga[g] = a;
      }
#pragma unroll
      for (int r = 0; r < 8; ++r) {
        float iv = sigm(ga[0][r]), fv = sigm(ga[1][r]);
        float gv = tanhfast(ga[2][r]), ov = sigm(ga[3][r]);
        float cn = fv * cst[jt][r] + iv * gv;
        cst[jt][r] = cn;
        hC[jt][r] = ov * tanhfast(cn);
      }
    }

    // ---- C-layout -> A-layout f16 via LDS round-trip ----
#pragma unroll
    for (int jt = 0; jt < 4; ++jt)
#pragma unroll
      for (int r = 0; r < 8; ++r)
        ht[(r + sel * 8) * 64 + jt * 16 + n15] = (_Float16)hC[jt][r];
    asm volatile("s_wait_dscnt 0" ::: "memory");
    loadA(ht, row, sel, aLo, aHi);
    asm volatile("s_wait_dscnt 0" ::: "memory");

    // ---- social pooling: per-lane masked A, per-cell W1 block ----
    v8f s1[4];
#pragma unroll
    for (int jt = 0; jt < 4; ++jt) {
      float bv = b1s[jt * 16 + n15];
#pragma unroll
      for (int r = 0; r < 8; ++r) s1[jt][r] = bv;
    }
#pragma unroll 1
    for (int cell = 0; cell < GCELLS; ++cell) {
      bool p = (myidx == cell);
      if (__builtin_amdgcn_ballot_w32(p) == 0u) continue;  // wave-uniform skip
      V16 mLo, mHi;
#pragma unroll
      for (int i = 0; i < 8; ++i) {
        mLo.u[i] = p ? aLo.u[i] : 0u;
        mHi.u[i] = p ? aHi.u[i] : 0u;
      }
#pragma unroll
      for (int jt = 0; jt < 4; ++jt) {
        int fb = cell * 8 + jt * 2;
        s1[jt] = __builtin_amdgcn_wmma_f32_16x16x32_f16(
            false, mLo.v, false, ldsFrag(w1f, fb + 0, lane), (short)0, s1[jt], false, false);
        s1[jt] = __builtin_amdgcn_wmma_f32_16x16x32_f16(
            false, mHi.v, false, ldsFrag(w1f, fb + 1, lane), (short)0, s1[jt], false, false);
      }
    }

    // ---- relu -> A frags -> second MLP layer ----
#pragma unroll
    for (int jt = 0; jt < 4; ++jt)
#pragma unroll
      for (int r = 0; r < 8; ++r)
        ht[(r + sel * 8) * 64 + jt * 16 + n15] = (_Float16)fmaxf(s1[jt][r], 0.0f);
    asm volatile("s_wait_dscnt 0" ::: "memory");
    V16 rLo, rHi;
    loadA(ht, row, sel, rLo, rHi);
    asm volatile("s_wait_dscnt 0" ::: "memory");

    v8f cmb[4];
#pragma unroll
    for (int jt = 0; jt < 4; ++jt) {
      float bv = b2s[jt * 16 + n15];
      v8f a;
#pragma unroll
      for (int r = 0; r < 8; ++r) a[r] = bv;
      a = __builtin_amdgcn_wmma_f32_16x16x32_f16(
          false, rLo.v, false, ldsFrag(w2f, jt * 2 + 0, lane), (short)0, a, false, false);
      a = __builtin_amdgcn_wmma_f32_16x16x32_f16(
          false, rHi.v, false, ldsFrag(w2f, jt * 2 + 1, lane), (short)0, a, false, false);
#pragma unroll
      for (int r = 0; r < 8; ++r) cmb[jt][r] = hC[jt][r] + a[r];
    }

    // ---- output projection (N padded 2->16) ----
#pragma unroll
    for (int jt = 0; jt < 4; ++jt)
#pragma unroll
      for (int r = 0; r < 8; ++r)
        ht[(r + sel * 8) * 64 + jt * 16 + n15] = (_Float16)cmb[jt][r];
    asm volatile("s_wait_dscnt 0" ::: "memory");
    V16 cLo, cHi;
    loadA(ht, row, sel, cLo, cHi);
    asm volatile("s_wait_dscnt 0" ::: "memory");

    v8f oacc;
#pragma unroll
    for (int r = 0; r < 8; ++r) oacc[r] = (n15 < 2) ? bouts[n15] : 0.0f;
    oacc = __builtin_amdgcn_wmma_f32_16x16x32_f16(
        false, cLo.v, false, ldsFrag(woutf, 0, lane), (short)0, oacc, false, false);
    oacc = __builtin_amdgcn_wmma_f32_16x16x32_f16(
        false, cHi.v, false, ldsFrag(woutf, 1, lane), (short)0, oacc, false, false);

    // stage through LDS (pos buffer is dead here) for one coalesced 128B store
    if (n15 < 2) {
#pragma unroll
      for (int r = 0; r < 8; ++r)
        post[(r + sel * 8) * 2 + n15] = oacc[r];
    }
    asm volatile("s_wait_dscnt 0" ::: "memory");
    out[((size_t)t * BATCH + agent0) * 2 + lane] = post[lane];
  }
}

// ---------------------------------------------------------------------------
extern "C" void kernel_launch(void* const* d_in, const int* in_sizes, int n_in,
                              void* d_out, int out_size, void* d_ws, size_t ws_size,
                              hipStream_t stream) {
  const float* traj = (const float*)d_in[0];
  const float* Wih  = (const float*)d_in[1];
  const float* Whh  = (const float*)d_in[2];
  const float* bih  = (const float*)d_in[3];
  const float* bhh  = (const float*)d_in[4];
  const float* W1   = (const float*)d_in[5];
  const float* b1   = (const float*)d_in[6];
  const float* W2   = (const float*)d_in[7];
  const float* b2   = (const float*)d_in[8];
  const float* Wout = (const float*)d_in[9];
  const float* bout = (const float*)d_in[10];
  unsigned* ws = (unsigned*)d_ws;
  float* out = (float*)d_out;

  minmax_init<<<1, 64, 0, stream>>>(ws);
  dim3 rg(32, TSTEPS);
  minmax_reduce<<<rg, 256, 0, stream>>>(traj, ws);
  social_lstm_main<<<BATCH / AGENTS_PER_BLOCK, BLOCK_THREADS, LDS_BYTES, stream>>>(
      traj, Wih, Whh, bih, bhh, W1, b1, W2, b2, Wout, bout, ws, out);
}
